// V_Basic_52123723105092
// MI455X (gfx1250) — compile-verified, matrix operand order checked by reading, not run on hardware
//
#include <hip/hip_runtime.h>

#define D_DIM      128
#define LDS_STRIDE 136      // 128 + 8 halves padding -> conflict-free b128 LDS reads
#define TILE_ROWS  32
#define NN_NODES   50000
#define NN_EDGES   800000
#define NEG_SLOPE  0.2f

typedef _Float16 v8h  __attribute__((ext_vector_type(8)));
typedef _Float16 v16h __attribute__((ext_vector_type(16)));
typedef float    v8f  __attribute__((ext_vector_type(8)));

// Fragment load: 16 halves = two contiguous 16B chunks at k and k+16
// (CDNA5 16-bit A/B layout: lanes 0-15 hold K=[0..7]+[16..23], lanes 16-31 K=[8..15]+[24..31])
__device__ __forceinline__ v16h ld_frag(const _Float16* p) {
  v16h r;
  *(v8h*)&r       = *(const v8h*)p;
  *((v8h*)&r + 1) = *(const v8h*)(p + 16);
  return r;
}

// One wave computes a 16x32 slice of (A @ W^T): A rows from LDS, W rows (== B columns) from global f16.
__device__ __forceinline__ void gemm_32cols(const _Float16* __restrict__ sA,
                                            const _Float16* __restrict__ W,
                                            int rrow, int col0, int kbase,
                                            v8f& acc0, v8f& acc1) {
#pragma unroll
  for (int k0 = 0; k0 < D_DIM; k0 += 32) {
    v16h a  = ld_frag(sA + rrow * LDS_STRIDE + k0 + kbase);
    v16h b0 = ld_frag(W + col0 * D_DIM + k0 + kbase);
    v16h b1 = ld_frag(W + (col0 + 16) * D_DIM + k0 + kbase);
    acc0 = __builtin_amdgcn_wmma_f32_16x16x32_f16(false, a, false, b0, (short)0, acc0, false, false);
    acc1 = __builtin_amdgcn_wmma_f32_16x16x32_f16(false, a, false, b1, (short)0, acc1, false, false);
  }
}

__global__ __launch_bounds__(256) void zero_kernel(float4* __restrict__ p, int n4) {
  int i = blockIdx.x * blockDim.x + threadIdx.x;
  if (i < n4) p[i] = make_float4(0.f, 0.f, 0.f, 0.f);
}

__global__ __launch_bounds__(256) void cvt_weights_kernel(
    const float* __restrict__ w0, const float* __restrict__ w1,
    const float* __restrict__ w2, const float* __restrict__ w3,
    const float* __restrict__ w4, const float* __restrict__ w5,
    _Float16* __restrict__ o) {
  const int i = blockIdx.x * 256 + threadIdx.x;   // 6 * 16384 total elements
  const int m = i >> 14;
  const int j = i & 16383;
  const float* srcs[6] = {w0, w1, w2, w3, w4, w5};
  o[i] = (_Float16)srcs[m][j];
}

// Vp = lrelu(lrelu(V) @ Wp1^T + bp1) @ Wp2^T + bp2
__global__ __launch_bounds__(256) void node_kernel(
    const float* __restrict__ V, const _Float16* __restrict__ W16,
    const float* __restrict__ bp1, const float* __restrict__ bp2,
    float* __restrict__ Vp) {
  __shared__ _Float16 sA[TILE_ROWS * LDS_STRIDE];
  __shared__ _Float16 sB[TILE_ROWS * LDS_STRIDE];
  const int t  = threadIdx.x;
  const int r0 = blockIdx.x * TILE_ROWS;

  { // stage 0: load V tile, apply leaky-relu, convert to f16 in LDS
    const int row  = t >> 3;
    const int cb   = (t & 7) * 16;
    const int grow = r0 + row;
#pragma unroll
    for (int c = 0; c < 16; c += 4) {
      float4 v = make_float4(0.f, 0.f, 0.f, 0.f);
      if (grow < NN_NODES) v = *(const float4*)(V + grow * D_DIM + cb + c);
      v.x = v.x > 0.f ? v.x : v.x * NEG_SLOPE;
      v.y = v.y > 0.f ? v.y : v.y * NEG_SLOPE;
      v.z = v.z > 0.f ? v.z : v.z * NEG_SLOPE;
      v.w = v.w > 0.f ? v.w : v.w * NEG_SLOPE;
      _Float16* p = sA + row * LDS_STRIDE + cb + c;
      p[0] = (_Float16)v.x; p[1] = (_Float16)v.y;
      p[2] = (_Float16)v.z; p[3] = (_Float16)v.w;
    }
  }
  __syncthreads();

  const _Float16* Wp1 = W16;
  const _Float16* Wp2 = W16 + 16384;
  const int wave = t >> 5, lane = t & 31;
  const int rt   = wave >> 2;            // 0..1 : 16-row tile
  const int ctb  = (wave & 3) * 32;      // 0,32,64,96 : 32-col tile
  const int half = (lane & 16) ? 8 : 0;  // K-half select (A/B) and M offset (C/D)
  const int rrow = rt * 16 + (lane & 15);
  const int colA = ctb + (lane & 15);
  const int colB = colA + 16;

  // stage 1: h1 = lrelu(sA @ Wp1^T + bp1) -> sB (f16)
  v8f a0 = {}, a1 = {};
  gemm_32cols(sA, Wp1, rrow, colA, half, a0, a1);
  {
    const float b0 = bp1[colA], b1 = bp1[colB];
#pragma unroll
    for (int r = 0; r < 8; r++) {
      const int m = rt * 16 + r + half;
      float x0 = a0[r] + b0; x0 = x0 > 0.f ? x0 : x0 * NEG_SLOPE;
      float x1 = a1[r] + b1; x1 = x1 > 0.f ? x1 : x1 * NEG_SLOPE;
      sB[m * LDS_STRIDE + colA] = (_Float16)x0;
      sB[m * LDS_STRIDE + colB] = (_Float16)x1;
    }
  }
  __syncthreads();

  // stage 2: Vp = sB @ Wp2^T + bp2 -> global f32
  v8f c0 = {}, c1 = {};
  gemm_32cols(sB, Wp2, rrow, colA, half, c0, c1);
  {
    const float b0 = bp2[colA], b1 = bp2[colB];
#pragma unroll
    for (int r = 0; r < 8; r++) {
      const int grow = r0 + rt * 16 + r + half;
      if (grow < NN_NODES) {
        Vp[grow * D_DIM + colA] = c0[r] + b0;
        Vp[grow * D_DIM + colB] = c1[r] + b1;
      }
    }
  }
}

// x = relu(E@WA1^T+bA1)@WA2^T+bA2 ; msg = sigmoid(x@WB^T+bB)*Vp[src] + (x@WC^T+bC) ; out[dst] += msg
__global__ __launch_bounds__(256) void edge_kernel(
    const float* __restrict__ E, const int* __restrict__ src, const int* __restrict__ dst,
    const _Float16* __restrict__ W16,
    const float* __restrict__ bA1, const float* __restrict__ bA2,
    const float* __restrict__ bB,  const float* __restrict__ bC,
    const float* __restrict__ Vp,  float* __restrict__ out) {
  __shared__ _Float16 sA[TILE_ROWS * LDS_STRIDE];
  __shared__ _Float16 sB[TILE_ROWS * LDS_STRIDE];
  __shared__ int sSrc[TILE_ROWS];
  __shared__ int sDst[TILE_ROWS];
  const int t  = threadIdx.x;
  const int e0 = blockIdx.x * TILE_ROWS;   // N_EDGES % 32 == 0

  { // stage 0: E tile -> f16 LDS; src/dst -> LDS
    const int row = t >> 3;
    const int cb  = (t & 7) * 16;
#pragma unroll
    for (int c = 0; c < 16; c += 4) {
      float4 v = *(const float4*)(E + (e0 + row) * D_DIM + cb + c);
      _Float16* p = sA + row * LDS_STRIDE + cb + c;
      p[0] = (_Float16)v.x; p[1] = (_Float16)v.y;
      p[2] = (_Float16)v.z; p[3] = (_Float16)v.w;
    }
    if (t < TILE_ROWS) { sSrc[t] = src[e0 + t]; sDst[t] = dst[e0 + t]; }
  }
  __syncthreads();

  const _Float16* WA1 = W16 + 2 * 16384;
  const _Float16* WA2 = W16 + 3 * 16384;
  const _Float16* WBm = W16 + 4 * 16384;
  const _Float16* WCm = W16 + 5 * 16384;
  const int wave = t >> 5, lane = t & 31;
  const int rt   = wave >> 2;
  const int ctb  = (wave & 3) * 32;
  const int half = (lane & 16) ? 8 : 0;
  const int rrow = rt * 16 + (lane & 15);
  const int colA = ctb + (lane & 15);
  const int colB = colA + 16;

  // stage 1: h = relu(E @ WA1^T + bA1) -> sB
  v8f a0 = {}, a1 = {};
  gemm_32cols(sA, WA1, rrow, colA, half, a0, a1);
  {
    const float b0 = bA1[colA], b1 = bA1[colB];
#pragma unroll
    for (int r = 0; r < 8; r++) {
      const int m = rt * 16 + r + half;
      sB[m * LDS_STRIDE + colA] = (_Float16)fmaxf(a0[r] + b0, 0.f);
      sB[m * LDS_STRIDE + colB] = (_Float16)fmaxf(a1[r] + b1, 0.f);
    }
  }
  __syncthreads();

  // stage 2: x = h @ WA2^T + bA2 -> sA (overwrites E tile; safe: nothing reads sA past barrier)
  v8f c0 = {}, c1 = {};
  gemm_32cols(sB, WA2, rrow, colA, half, c0, c1);
  {
    const float b0 = bA2[colA], b1 = bA2[colB];
#pragma unroll
    for (int r = 0; r < 8; r++) {
      const int m = rt * 16 + r + half;
      sA[m * LDS_STRIDE + colA] = (_Float16)(c0[r] + b0);
      sA[m * LDS_STRIDE + colB] = (_Float16)(c1[r] + b1);
    }
  }
  __syncthreads();

  // stage 3: scale = sigmoid(x@WB^T+bB), shift = x@WC^T+bC (both kept in registers)
  v8f s0 = {}, s1 = {}, h0 = {}, h1 = {};
  gemm_32cols(sA, WBm, rrow, colA, half, s0, s1);
  gemm_32cols(sA, WCm, rrow, colA, half, h0, h1);
  {
    const float bB0 = bB[colA], bB1 = bB[colB];
    const float bC0 = bC[colA], bC1 = bC[colB];
#pragma unroll
    for (int r = 0; r < 8; r++) {
      const int m = rt * 16 + r + half;
      const int s = sSrc[m];
      const int d = sDst[m];
      const float sc0 = 1.f / (1.f + __expf(-(s0[r] + bB0)));
      const float sc1 = 1.f / (1.f + __expf(-(s1[r] + bB1)));
      const float m0 = sc0 * Vp[s * D_DIM + colA] + (h0[r] + bC0);
      const float m1 = sc1 * Vp[s * D_DIM + colB] + (h1[r] + bC1);
      atomicAdd(out + d * D_DIM + colA, m0);   // global_atomic_add_f32, L2-resident target
      atomicAdd(out + d * D_DIM + colB, m1);
    }
  }
}

extern "C" void kernel_launch(void* const* d_in, const int* in_sizes, int n_in,
                              void* d_out, int out_size, void* d_ws, size_t ws_size,
                              hipStream_t stream) {
  const float* V   = (const float*)d_in[0];
  const float* E   = (const float*)d_in[1];
  const int*   src = (const int*)d_in[2];
  const int*   dst = (const int*)d_in[3];
  const float* Wp1 = (const float*)d_in[4];
  const float* bp1 = (const float*)d_in[5];
  const float* Wp2 = (const float*)d_in[6];
  const float* bp2 = (const float*)d_in[7];
  const float* WA1 = (const float*)d_in[8];
  const float* bA1 = (const float*)d_in[9];
  const float* WA2 = (const float*)d_in[10];
  const float* bA2 = (const float*)d_in[11];
  const float* WB  = (const float*)d_in[12];
  const float* bB  = (const float*)d_in[13];
  const float* WC  = (const float*)d_in[14];
  const float* bC  = (const float*)d_in[15];
  float* out = (float*)d_out;

  // workspace layout: [ 6 x 128x128 f16 weights | Vp f32 50000x128 ]
  _Float16* W16 = (_Float16*)d_ws;
  float* Vp = (float*)((char*)d_ws + (size_t)6 * 16384 * sizeof(_Float16));

  const int n4 = NN_NODES * D_DIM / 4;
  zero_kernel<<<(n4 + 255) / 256, 256, 0, stream>>>((float4*)out, n4);
  cvt_weights_kernel<<<(6 * 16384) / 256, 256, 0, stream>>>(Wp1, Wp2, WA1, WA2, WB, WC, W16);
  node_kernel<<<(NN_NODES + TILE_ROWS - 1) / TILE_ROWS, 256, 0, stream>>>(V, W16, bp1, bp2, Vp);
  edge_kernel<<<NN_EDGES / TILE_ROWS, 256, 0, stream>>>(E, src, dst, W16, bA1, bA2, bB, bC, Vp, out);
}
